// TripletInBatchNegCoSentLoss_75943611728627
// MI455X (gfx1250) — compile-verified, matrix operand order checked by reading, not run on hardware
//
#include <hip/hip_runtime.h>
#include <hip/hip_bf16.h>

typedef __attribute__((ext_vector_type(16))) _Float16 v16h;
typedef __attribute__((ext_vector_type(8)))  _Float16 v8h;
typedef __attribute__((ext_vector_type(8)))  float    v8f;
typedef __attribute__((ext_vector_type(4)))  int      v4i;

#define D_DIM 256
#define LOG2E 1.4426950408889634f
#define KSCALE (LOG2E * 20.0f)   // log2(e) / T, T = 0.05

// LDS B-tile: 16 rows x 256 halves, row padded by 8 halves (16B) so the 16
// fragment-reading lanes (stride 132 dwords == 4 mod 64 banks) are conflict-free.
#define ROWH  264
#define TILEH (16 * ROWH)

#if defined(__gfx1250__) && __has_builtin(__builtin_amdgcn_global_load_async_to_lds_b128) && \
    __has_builtin(__builtin_amdgcn_s_wait_asynccnt)
#define USE_ASYNC 1
#else
#define USE_ASYNC 0
#endif

#if defined(__gfx1250__) && __has_builtin(__builtin_amdgcn_exp2f)
#define EXP2F(x) __builtin_amdgcn_exp2f(x)   // raw v_exp_f32; |exponent| <= ~58, safe
#else
#define EXP2F(x) exp2f(x)
#endif

// ---- Kernel 1: row L2-normalize fp32 -> f16 (one block per row, 256 thr) ----
__global__ void rownorm_f16_kernel(const float* __restrict__ x,
                                   _Float16* __restrict__ y) {
  const int row = blockIdx.x;
  const int t   = threadIdx.x;                 // D_DIM == blockDim.x == 256
  const size_t base = (size_t)row * D_DIM;
  float v  = x[base + t];
  float ss = v * v;
#pragma unroll
  for (int m = 16; m >= 1; m >>= 1) ss += __shfl_xor(ss, m, 32);
  __shared__ float red[8];
  if ((t & 31) == 0) red[t >> 5] = ss;
  __syncthreads();
  float tot = red[0] + red[1] + red[2] + red[3] +
              red[4] + red[5] + red[6] + red[7];
  float scale = 1.0f / fmaxf(sqrtf(tot), 1e-8f);   // torch clamps norm at 1e-8
  y[base + t] = (_Float16)(v * scale);
}

// ---- Kernel 2: sim_pos[i] = dot(e_h[i], p_h[i]) with fp32 accumulation ----
__global__ void simpos_kernel(const _Float16* __restrict__ e,
                              const _Float16* __restrict__ p,
                              float* __restrict__ sp) {
  const int row = blockIdx.x;
  const int t   = threadIdx.x;
  const size_t base = (size_t)row * D_DIM;
  float v = (float)e[base + t] * (float)p[base + t];
#pragma unroll
  for (int m = 16; m >= 1; m >>= 1) v += __shfl_xor(v, m, 32);
  __shared__ float red[8];
  if ((t & 31) == 0) red[t >> 5] = v;
  __syncthreads();
  if (t == 0)
    sp[row] = red[0] + red[1] + red[2] + red[3] +
              red[4] + red[5] + red[6] + red[7];
}

// ---- Kernel 3: fused WMMA GEMM + row-wise sum of exp2((s - sp) * KSCALE) ----
// 8 waves/block, each wave owns a 16-row panel (A frags in registers).
// The block shares each 16-column B tile through a double-buffered LDS stage
// filled with global_load_async_to_lds_b128 (ASYNCcnt), 8x L2-traffic reuse.
__global__ void __launch_bounds__(256)
wmma_rowlse_kernel(const _Float16* __restrict__ A,
                   const _Float16* __restrict__ B,
                   const float* __restrict__ sim_pos,
                   float* __restrict__ rowsum,
                   int N) {
  __shared__ __align__(16) _Float16 ldsB[2 * TILEH];

  const int t    = threadIdx.x;
  const int lane = t & 31;
  const int wave = t >> 5;
  const int i0   = (blockIdx.x * 8 + wave) * 16;
  const int half = lane >> 4;   // lane-half: selects K sub-ranges & C row group
  const int l16  = lane & 15;

  // Stage B rows [jt, jt+16) into ldsB buffer `buf`: 512 x 16B segments,
  // each of the 256 threads moves two segments.
  auto stage = [&](int jt, int buf) {
#pragma unroll
    for (int s2 = 0; s2 < 2; ++s2) {
      const int tr  = t + s2 * 256;
      const int row = tr >> 5;        // 0..15
      const int seg = tr & 31;        // 16B segment within the 512B row
      const _Float16* g = B + (size_t)(jt + row) * D_DIM + seg * 8;
      _Float16*       l = ldsB + buf * TILEH + row * ROWH + seg * 8;
      __builtin_prefetch(g + 16 * D_DIM, 0, 0);   // global_prefetch_b8 (speculative)
#if USE_ASYNC
      __builtin_amdgcn_global_load_async_to_lds_b128(
          (__attribute__((address_space(1))) v4i*)g,
          (__attribute__((address_space(3))) v4i*)l, 0, 0);
#else
      *(v8h*)l = *(const v8h*)g;
#endif
    }
  };

  // Preload A fragments for all 8 K-chunks (K = 256 = 8 * 32).
  // ISA 16-bit A layout: lane holds M = lane%16; per half h, VGPR0-3 hold
  // K = h*8 .. h*8+7 and VGPR4-7 hold K = h*8+16 .. h*8+23 (2 halves packed).
  const _Float16* Arow = A + (size_t)(i0 + l16) * D_DIM;
  v16h afrag[8];
#pragma unroll
  for (int kc = 0; kc < 8; ++kc) {
    const int kb = kc * 32 + half * 8;
    v8h lo = *(const v8h*)(Arow + kb);        // global_load_b128
    v8h hi = *(const v8h*)(Arow + kb + 16);
#pragma unroll
    for (int i = 0; i < 8; ++i) { afrag[kc][i] = lo[i]; afrag[kc][i + 8] = hi[i]; }
  }

  // C/D layout: slot r in lanes 0-15 is row i0+r; in lanes 16-31 it is i0+8+r.
  float spb[8], acc[8];
#pragma unroll
  for (int r = 0; r < 8; ++r) {
    spb[r] = sim_pos[i0 + half * 8 + r] * KSCALE;
    acc[r] = 0.0f;
  }

  stage(0, 0);                       // prologue: fill buffer 0

  for (int jt = 0; jt < N; jt += 16) {
    const int buf = (jt >> 4) & 1;
    if (jt + 16 < N) {
      stage(jt + 16, buf ^ 1);       // overwrites buffer consumed 2 tiles ago
#if USE_ASYNC
      __builtin_amdgcn_s_wait_asynccnt(2);   // current tile's 2 async ops done
#endif
    } else {
#if USE_ASYNC
      __builtin_amdgcn_s_wait_asynccnt(0);
#endif
    }
    __syncthreads();                 // tile visible to all 8 waves

    const _Float16* Brow = ldsB + buf * TILEH + l16 * ROWH;  // column jt + l16

    // Load ALL 8 B fragments into distinct registers first -> one ds clause +
    // single s_wait_dscnt, instead of 8 exposed load->wmma round trips.
    v16h bfr[8];
#pragma unroll
    for (int kc = 0; kc < 8; ++kc) {
      const int kb = kc * 32 + half * 8;
      v8h lo = *(const v8h*)(Brow + kb);      // ds_load_b128, conflict-free
      v8h hi = *(const v8h*)(Brow + kb + 16);
#pragma unroll
      for (int i = 0; i < 8; ++i) { bfr[kc][i] = lo[i]; bfr[kc][i + 8] = hi[i]; }
    }

    v8f c = {};
#pragma unroll
    for (int kc = 0; kc < 8; ++kc) {
      // (neg_a, A, neg_b, B, c_mod, C, reuse_a, reuse_b)
      c = __builtin_amdgcn_wmma_f32_16x16x32_f16(false, afrag[kc], false, bfr[kc],
                                                 (short)0, c, false, false);
    }

    // Exponent bounded by ~|2| * KSCALE ~= 58 -> raw v_exp_f32 is safe.
#pragma unroll
    for (int r = 0; r < 8; ++r)
      acc[r] += EXP2F(c[r] * KSCALE - spb[r]);

    __syncthreads();                 // all waves done with `buf` before refill
  }

  // Sum over the 16 columns held across each lane-half (masks < 16 stay in-half).
#pragma unroll
  for (int r = 0; r < 8; ++r) {
#pragma unroll
    for (int m = 8; m >= 1; m >>= 1) acc[r] += __shfl_xor(acc[r], m, 32);
  }
  if (l16 == 0) {
#pragma unroll
    for (int r = 0; r < 8; ++r) rowsum[i0 + half * 8 + r] = acc[r];
  }
}

// ---- Kernel 4: out = mean(log(1 + r1)) + mean(log(r2)) ----
__global__ void finalize_kernel(const float* __restrict__ r1,
                                const float* __restrict__ r2,
                                float* __restrict__ out, int N) {
  const int t = threadIdx.x;   // single block of 256
  float s = 0.0f;
  for (int i = t; i < N; i += 256)
    s += log2f(1.0f + r1[i]) + log2f(r2[i]);
#pragma unroll
  for (int m = 16; m >= 1; m >>= 1) s += __shfl_xor(s, m, 32);
  __shared__ float red[8];
  if ((t & 31) == 0) red[t >> 5] = s;
  __syncthreads();
  if (t == 0) {
    float tot = red[0] + red[1] + red[2] + red[3] +
                red[4] + red[5] + red[6] + red[7];
    out[0] = tot * (0.69314718055994531f / (float)N);  // ln2 converts log2 -> ln
  }
}

extern "C" void kernel_launch(void* const* d_in, const int* in_sizes, int n_in,
                              void* d_out, int out_size, void* d_ws, size_t ws_size,
                              hipStream_t stream) {
  const float* e = (const float*)d_in[0];
  const float* p = (const float*)d_in[1];
  const float* n = (const float*)d_in[2];
  const int N = in_sizes[0] / D_DIM;   // 8192

  char* ws = (char*)d_ws;
  const size_t matBytes = (size_t)N * D_DIM * sizeof(_Float16);
  _Float16* e_h = (_Float16*)ws; ws += matBytes;
  _Float16* p_h = (_Float16*)ws; ws += matBytes;
  _Float16* n_h = (_Float16*)ws; ws += matBytes;
  float* sp = (float*)ws; ws += (size_t)N * sizeof(float);
  float* r1 = (float*)ws; ws += (size_t)N * sizeof(float);
  float* r2 = (float*)ws;

  rownorm_f16_kernel<<<N, 256, 0, stream>>>(e, e_h);
  rownorm_f16_kernel<<<N, 256, 0, stream>>>(p, p_h);
  rownorm_f16_kernel<<<N, 256, 0, stream>>>(n, n_h);
  simpos_kernel<<<N, 256, 0, stream>>>(e_h, p_h, sp);

  // loss1: rows of e vs all of n; 512 panels, 8 waves/block -> N/128 blocks
  wmma_rowlse_kernel<<<N / 128, 256, 0, stream>>>(e_h, n_h, sp, r1, N);
  // swap pair loss: rows of p vs all of e (diagonal subtract == sim_pos)
  wmma_rowlse_kernel<<<N / 128, 256, 0, stream>>>(p_h, e_h, sp, r2, N);

  finalize_kernel<<<1, 256, 0, stream>>>(r1, r2, (float*)d_out, N);
}
